// ExpHydroM100_77025943486915
// MI455X (gfx1250) — matrix operand order repaired and checked by reading, not compiled, not run on hardware
//
#include <hip/hip_runtime.h>

// ============================================================================
// ExpHydro M100: sequential Euler scan (MLP per step) + batched MLP readout.
// CDNA5 strategy:
//  - prep_weights: W1/W2 f32 -> f16, pre-swizzled into the exact WMMA
//    B-operand (32x16, KxN) per-lane register layout, stored in d_ws.
//  - scan_kernel: 1 workgroup (8 waves), both weight matrices LDS-resident
//    (2 x 128KB f16). Each 256x256 matvec runs as v_wmma_f32_16x16x32_f16
//    with the activation vector replicated into all 16 A rows. Activations
//    are kept in LDS as *A-swizzled f16*, so each K-step's A operand is one
//    aligned 32B ds_load (no f32 loads / cvt on the serial critical path).
//  - final_mlp: 229 blocks x 1 wave; two 16x256x256 WMMA GEMM layers.
// ============================================================================

#define TT 3650
#define HH 256

typedef __attribute__((ext_vector_type(16))) _Float16 v16h;
typedef __attribute__((ext_vector_type(8)))  float    v8f;

__device__ __forceinline__ float stepf(float x) {
    return (tanhf(5.0f * x) + 1.0f) * 0.5f;
}

// A-swizzle permutation within a 32-element K chunk:
// positions [0..15] hold K {0..7,16..23}  (read by lanes 0-15)
// positions [16..31] hold K {8..15,24..31} (read by lanes 16-31)
__device__ __forceinline__ int aswz_pos(int col) {
    int r = col & 31, chunk = col >> 5;
    int p = (r < 8) ? r : (r < 16) ? r + 8 : (r < 24) ? r - 8 : r;
    return chunk * 32 + p;
}

// ---------------------------------------------------------------------------
// Pre-swizzle W1/W2 (256x256 f32 row-major, y = h @ W so W[k][n]) into f16
// WMMA B-operand blocks: block (nt,kt) is 32 lanes x 16 halves, contiguous.
// element index e = ((nt*8+kt)*32 + lane)*16 + i
//   maps to  W[(kt*32 + (lane<16?0:8) + (i<8?i:i+8)) * 256 + nt*16 + (lane&15)]
// ---------------------------------------------------------------------------
__global__ void __launch_bounds__(256) prep_weights(
    const float* __restrict__ W1, const float* __restrict__ W2,
    _Float16* __restrict__ w1h, _Float16* __restrict__ w2h) {
    int e = blockIdx.x * blockDim.x + threadIdx.x;   // 0 .. 131071
    int mat  = e >> 16;
    int r    = e & 65535;
    int i    = r & 15;
    int lane = (r >> 4) & 31;
    int ktnt = r >> 9;            // 0..127
    int kt   = ktnt & 7;
    int nt   = ktnt >> 3;
    int K = kt * 32 + ((lane < 16) ? 0 : 8) + ((i < 8) ? i : i + 8);
    int N = nt * 16 + (lane & 15);
    const float* src = mat ? W2 : W1;
    _Float16*    dst = mat ? w2h : w1h;
    dst[r] = (_Float16)src[K * 256 + N];
}

// ---------------------------------------------------------------------------
// One 256->256 tanh layer as WMMA matvec. A has all 16 rows equal to h
// (every lane reads the shared swizzled f16 vector), so each lane's D VGPR0
// is a valid output. Wave w owns column tiles 2w / 2w+1; lanes 0-15 commit
// tile 2w, lanes 16-31 tile 2w+1.
// ---------------------------------------------------------------------------
template <bool F16OUT>
__device__ __forceinline__ void wmma_matvec_layer(
    const _Float16* __restrict__ Wblk,   // LDS, swizzled B blocks
    const _Float16* __restrict__ hinh,   // LDS, 256 halves, A-swizzled
    const float*    __restrict__ bias,   // LDS, 256 f32
    _Float16*       __restrict__ houth,  // LDS, A-swizzled f16 out (if F16OUT)
    float*          __restrict__ houtf,  // LDS, f32 out (if !F16OUT)
    int w, int lane) {
    const int nt0 = 2 * w, nt1 = 2 * w + 1;
    v8f acc0 = {0.f,0.f,0.f,0.f,0.f,0.f,0.f,0.f};
    v8f acc1 = {0.f,0.f,0.f,0.f,0.f,0.f,0.f,0.f};
    const _Float16* abase = hinh + ((lane < 16) ? 0 : 16);
#pragma unroll
    for (int kt = 0; kt < 8; ++kt) {
        v16h a   = *(const v16h*)(abase + kt * 32);
        v16h bv0 = *(const v16h*)(Wblk + ((nt0 * 8 + kt) * 32 + lane) * 16);
        v16h bv1 = *(const v16h*)(Wblk + ((nt1 * 8 + kt) * 32 + lane) * 16);
        acc0 = __builtin_amdgcn_wmma_f32_16x16x32_f16(false, a, false, bv0,
                                                      (short)0, acc0, false, false);
        acc1 = __builtin_amdgcn_wmma_f32_16x16x32_f16(false, a, false, bv1,
                                                      (short)0, acc1, false, false);
    }
    int col; float v;
    if (lane < 16) { col = nt0 * 16 + lane;        v = acc0[0]; }
    else           { col = nt1 * 16 + (lane - 16); v = acc1[0]; }
    float r = tanhf(v + bias[col]);
    if (F16OUT) houth[aswz_pos(col)] = (_Float16)r;
    else        houtf[col] = r;
}

// ---------------------------------------------------------------------------
// Sequential scan: one workgroup, weights LDS-resident.
// out layout: q[0,T) | s_snow[T,2T) | s_water[2T,3T)
//
// LDS layout (bytes from base):
//   0      W1s   131072   (f16 B blocks)
//   131072 W2s   131072
//   262144 W0s     4096   (f32 4x256)
//   266240 b0s     1024
//   267264 b1s     1024
//   268288 b2s     1024
//   269312 Wos     5120   (f32 256x5)
//   274432 bos       32
//   274464 h3      1024   (f32)
//   275488 h1h      512   (f16 A-swizzled)
//   276000 h2h      512   (f16 A-swizzled)
//   276512 sst       32
//   total 276544
// ---------------------------------------------------------------------------
#define SCAN_SMEM_BYTES 276544

__global__ void __launch_bounds__(256) scan_kernel(
    const float* __restrict__ inputs,  // (T,5): d0 d1 prcp tmean time
    const float* __restrict__ dayl,
    const float* __restrict__ W0, const float* __restrict__ b0,
    const float* __restrict__ b1, const float* __restrict__ b2,
    const float* __restrict__ Wout, const float* __restrict__ bout,
    const _Float16* __restrict__ w1h, const _Float16* __restrict__ w2h,
    float* __restrict__ out) {
    extern __shared__ __align__(32) char smem[];
    _Float16* W1s = (_Float16*)smem;
    _Float16* W2s = (_Float16*)(smem + 131072);
    float*    W0s = (float*)(smem + 262144);
    float*    b0s = (float*)(smem + 266240);
    float*    b1s = (float*)(smem + 267264);
    float*    b2s = (float*)(smem + 268288);
    float*    Wos = (float*)(smem + 269312);
    float*    bos = (float*)(smem + 274432);
    float*    h3  = (float*)(smem + 274464);
    _Float16* h1h = (_Float16*)(smem + 275488);
    _Float16* h2h = (_Float16*)(smem + 276000);
    float*    sst = (float*)(smem + 276512);

    const int tid  = threadIdx.x;
    const int w    = tid >> 5;
    const int lane = tid & 31;

    // Stage weights: both swizzled matrices are contiguous in d_ws.
    {
        const float4* src = (const float4*)w1h;
        float4* dst = (float4*)smem;
        for (int i = tid; i < 16384; i += 256) dst[i] = src[i];   // 256KB
        for (int i = tid; i < 1024; i += 256) W0s[i] = W0[i];
        b0s[tid] = b0[tid]; b1s[tid] = b1[tid]; b2s[tid] = b2[tid];
        for (int i = tid; i < 1280; i += 256) Wos[i] = Wout[i];
        if (tid < 5) bos[tid] = bout[tid];
        if (tid == 0) {
            float s0 = inputs[0], s1 = inputs[1];
            sst[0] = s0; sst[1] = s1;
            out[TT] = s0;        // s_snow_nn[0]
            out[2 * TT] = s1;    // s_water_nn[0]
        }
    }
    __syncthreads();

    const int myswz = aswz_pos(tid);

    for (int t = 0; t < TT - 1; ++t) {
        const float s0 = sst[0], s1 = sst[1];
        const float p  = inputs[t * 5 + 2];
        const float tm = inputs[t * 5 + 3];

        // layer 0: 4 -> 256, written A-swizzled f16
        float z = s0 * W0s[tid] + s1 * W0s[256 + tid] +
                  p  * W0s[512 + tid] + tm * W0s[768 + tid] + b0s[tid];
        h1h[myswz] = (_Float16)tanhf(z);
        __syncthreads();

        wmma_matvec_layer<true >(W1s, h1h, b1s, h2h, (float*)nullptr, w, lane);
        __syncthreads();
        wmma_matvec_layer<false>(W2s, h2h, b2s, (_Float16*)nullptr, h3, w, lane);
        __syncthreads();

        // output layer (256 -> 5) + physics update, all inside wave 0
        if (w == 0) {
            float a0 = 0.f, a1 = 0.f, a2 = 0.f, a3 = 0.f, a4 = 0.f;
#pragma unroll
            for (int j = 0; j < 8; ++j) {
                int k = lane + 32 * j;
                float hv = h3[k];
                const float* wr = &Wos[k * 5];
                a0 += hv * wr[0]; a1 += hv * wr[1]; a2 += hv * wr[2];
                a3 += hv * wr[3]; a4 += hv * wr[4];
            }
#pragma unroll
            for (int off = 16; off > 0; off >>= 1) {
                a0 += __shfl_xor(a0, off, 32);
                a1 += __shfl_xor(a1, off, 32);
                a2 += __shfl_xor(a2, off, 32);
                a3 += __shfl_xor(a3, off, 32);
                a4 += __shfl_xor(a4, off, 32);
            }
            if (lane == 0) {
                float o0 = a0 + bos[0], o1 = a1 + bos[1], o2 = a2 + bos[2];
                float o3 = a3 + bos[3], o4 = a4 + bos[4];
                const float ld = dayl[t];
                float p_snow = fmaxf(sinhf(o0) * stepf(-tm), 0.f);
                float p_rain = fmaxf(sinhf(o1), 0.f);
                float melt   = fmaxf(stepf(s0) * sinhf(o2), 0.f);
                float et     = stepf(s1) * expf(o3) * ld;
                float q      = stepf(s1) * expf(o4);
                float ns0 = s0 + (p_snow - melt);
                float ns1 = s1 + (p_rain + melt - et - q);
                out[TT + t + 1]     = ns0;
                out[2 * TT + t + 1] = ns1;
                sst[0] = ns0; sst[1] = ns1;
            }
        }
        __syncthreads();
    }
}

// ---------------------------------------------------------------------------
// One 16x256 @ 256x256 tanh GEMM layer, single wave, WMMA, B from global
// pre-swizzled f16 (fully coalesced 32B/lane loads).
// ---------------------------------------------------------------------------
__device__ __forceinline__ void wmma_gemm_layer(
    const _Float16* __restrict__ Wblk,      // global, swizzled blocks
    const float (*__restrict__ hin)[256],   // LDS
    const float* __restrict__ bias,         // global
    float (*__restrict__ hout)[256],        // LDS
    int lane) {
    const int m  = lane & 15;
    const int kb = (lane < 16) ? 0 : 8;
    const int rowbase = (lane < 16) ? 0 : 8;
    for (int g = 0; g < 2; ++g) {
        v8f acc[8];
#pragma unroll
        for (int j = 0; j < 8; ++j) acc[j] = (v8f){0.f,0.f,0.f,0.f,0.f,0.f,0.f,0.f};
        for (int kt = 0; kt < 8; ++kt) {
            v16h a;
#pragma unroll
            for (int i = 0; i < 8; ++i) {
                a[i]     = (_Float16)hin[m][kt * 32 + kb + i];
                a[8 + i] = (_Float16)hin[m][kt * 32 + 16 + kb + i];
            }
#pragma unroll
            for (int j = 0; j < 8; ++j) {
                int nt = g * 8 + j;
                v16h bv = *(const v16h*)(Wblk + ((nt * 8 + kt) * 32 + lane) * 16);
                acc[j] = __builtin_amdgcn_wmma_f32_16x16x32_f16(false, a, false, bv,
                                                                (short)0, acc[j], false, false);
            }
        }
#pragma unroll
        for (int j = 0; j < 8; ++j) {
            int col = (g * 8 + j) * 16 + m;
            float bc = bias[col];
#pragma unroll
            for (int r = 0; r < 8; ++r)
                hout[rowbase + r][col] = tanhf(acc[j][r] + bc);
        }
    }
}

// ---------------------------------------------------------------------------
// Batched readout MLP: one wave per 16-row tile; q = mlp(x)[:,4].
// ---------------------------------------------------------------------------
__global__ void __launch_bounds__(32) final_mlp(
    const float* __restrict__ inputs,
    const float* __restrict__ W0, const float* __restrict__ b0,
    const float* __restrict__ b1, const float* __restrict__ b2,
    const float* __restrict__ Wout, const float* __restrict__ bout,
    const _Float16* __restrict__ w1h, const _Float16* __restrict__ w2h,
    float* __restrict__ out) {
    __shared__ float X[16][4];
    __shared__ float hA[16][256];
    __shared__ float hB[16][256];
    const int lane = threadIdx.x;
    const int r0 = blockIdx.x * 16;
    const float* ssnow  = out + TT;
    const float* swater = out + 2 * TT;

    if (lane < 16) {
        int r = r0 + lane;
        if (r < TT) {
            X[lane][0] = ssnow[r];
            X[lane][1] = swater[r];
            X[lane][2] = inputs[r * 5 + 2];
            X[lane][3] = inputs[r * 5 + 3];
        } else {
            X[lane][0] = X[lane][1] = X[lane][2] = X[lane][3] = 0.f;
        }
    }
    __syncthreads();

    // layer 0: 16x4 @ 4x256
    for (int idx = lane; idx < 4096; idx += 32) {
        int m = idx >> 8, c = idx & 255;
        hA[m][c] = tanhf(X[m][0] * W0[c] + X[m][1] * W0[256 + c] +
                         X[m][2] * W0[512 + c] + X[m][3] * W0[768 + c] + b0[c]);
    }
    __syncthreads();

    wmma_gemm_layer(w1h, hA, b1, hB, lane);
    __syncthreads();
    wmma_gemm_layer(w2h, hB, b2, hA, lane);
    __syncthreads();

    // readout: q = hA @ Wout[:,4] + bout[4]
    {
        int row = lane & 15, half = lane >> 4;
        float s = 0.f;
        for (int k = half * 128; k < half * 128 + 128; ++k)
            s += hA[row][k] * Wout[k * 5 + 4];
        s += __shfl_xor(s, 16, 32);
        int r = r0 + row;
        if (lane < 16 && r < TT) out[r] = s + bout[4];
    }
}

// ---------------------------------------------------------------------------
extern "C" void kernel_launch(void* const* d_in, const int* in_sizes, int n_in,
                              void* d_out, int out_size, void* d_ws, size_t ws_size,
                              hipStream_t stream) {
    const float* inputs = (const float*)d_in[0];
    const float* dayl   = (const float*)d_in[1];
    const float* W0     = (const float*)d_in[2];
    const float* b0     = (const float*)d_in[3];
    const float* W1     = (const float*)d_in[4];
    const float* b1     = (const float*)d_in[5];
    const float* W2     = (const float*)d_in[6];
    const float* b2     = (const float*)d_in[7];
    const float* Wout   = (const float*)d_in[8];
    const float* bout   = (const float*)d_in[9];
    float* out = (float*)d_out;

    _Float16* w1h = (_Float16*)d_ws;          // 128KB
    _Float16* w2h = w1h + 65536;              // 128KB

    // 1) swizzle weights to f16 WMMA B-layout (131072 elements total)
    prep_weights<<<512, 256, 0, stream>>>(W1, W2, w1h, w2h);

    // 2) sequential scan, weights LDS-resident
    scan_kernel<<<1, 256, SCAN_SMEM_BYTES, stream>>>(inputs, dayl, W0, b0, b1, b2,
                                                     Wout, bout, w1h, w2h, out);

    // 3) batched readout MLP (q for all T rows)
    final_mlp<<<(TT + 15) / 16, 32, 0, stream>>>(inputs, W0, b0, b1, b2,
                                                 Wout, bout, w1h, w2h, out);
}
